// DeltaOrderLoss_50757923504322
// MI455X (gfx1250) — compile-verified, matrix operand order checked by reading, not run on hardware
//
#include <hip/hip_runtime.h>
#include <math.h>

#define NROWS 256   // N = 2*128
#define DIMS  256
#define MCOLS 255   // N-1
#define DELTA_F 0.1f

typedef __attribute__((ext_vector_type(2)))  float    v2f;
typedef __attribute__((ext_vector_type(8)))  float    v8f;
typedef __attribute__((ext_vector_type(16))) _Float16 v16h;

#if defined(__has_builtin)
# if __has_builtin(__builtin_amdgcn_wmma_f32_16x16x4_f32)
#  define USE_WMMA_F32 1
# endif
#endif

// features [128,2,256]; logical stacked row i of F[256,256]:
//   i < 128  -> features[i,0,:]   at offset i*512
//   i >= 128 -> features[i-128,1,:] at offset (i-128)*512 + 256
__device__ __forceinline__ const float* frow(const float* f, int i) {
  return (i < 128) ? (f + (size_t)i * 512) : (f + (size_t)(i - 128) * 512 + 256);
}

// -------- kernel 1: row squared norms --------
__global__ void __launch_bounds__(NROWS) norms_kernel(const float* __restrict__ f,
                                                      float* __restrict__ norms) {
  const int i = threadIdx.x;
  const float* r = frow(f, i);
  float s = 0.0f;
  for (int d = 0; d < DIMS; ++d) { float v = r[d]; s = fmaf(v, v, s); }
  norms[i] = s;
}

// -------- kernel 2: Gram via WMMA, then z = sqrt(relu(ni+nj-2g)) --------
__global__ void __launch_bounds__(32) gram_z_kernel(const float* __restrict__ f,
                                                    const float* __restrict__ norms,
                                                    float* __restrict__ z) {
  const int tile = blockIdx.x;            // 256 tiles of 16x16
  const int rowBase = (tile >> 4) * 16;
  const int colBase = (tile & 15) * 16;
  const int lane = threadIdx.x;           // wave32, EXEC all ones
  const int m    = lane & 15;
  const int half = lane >> 4;

  const float* __restrict__ ra = frow(f, rowBase + m);
  const float* __restrict__ rb = frow(f, colBase + m);

  v8f acc = {};
#if defined(USE_WMMA_F32)
  // 32-bit A 16x4 layout (ISA 7.12.2): VGPR0 holds K = 2*half, VGPR1 holds K = 2*half+1.
  for (int kb = 0; kb < DIMS; kb += 4) {
    const int k0 = kb + 2 * half;
    v2f a, b;
    a.x = ra[k0]; a.y = ra[k0 + 1];
    b.x = rb[k0]; b.y = rb[k0 + 1];
    acc = __builtin_amdgcn_wmma_f32_16x16x4_f32(false, a, false, b,
                                                (short)0, acc, false, false);
  }
#else
  // fallback: fp16 WMMA 16x16x32 (codegen-confirmed builtin)
  for (int kb = 0; kb < DIMS; kb += 32) {
    v16h a, b;
    for (int e = 0; e < 16; ++e) {
      const int k = kb + 16 * (e >> 3) + 8 * half + (e & 7);
      a[e] = (_Float16)ra[k];
      b[e] = (_Float16)rb[k];
    }
    acc = __builtin_amdgcn_wmma_f32_16x16x32_f16(false, a, false, b,
                                                 (short)0, acc, false, false);
  }
#endif

  // C/D 16x16 f32 layout: VGPR v -> M = v + 8*half, N = lane&15
  const int j = colBase + m;
  const float nj = norms[j];
#pragma unroll
  for (int vg = 0; vg < 8; ++vg) {
    const int i = rowBase + vg + 8 * half;
    const float d2 = norms[i] + nj - 2.0f * acc[vg];
    z[i * NROWS + j] = sqrtf(fmaxf(d2, 0.0f));
  }
}

// -------- kernel 3: per-row loss terms --------
__global__ void __launch_bounds__(256) row_loss_kernel(const float* __restrict__ z,
                                                       const int* __restrict__ labels,
                                                       float* __restrict__ rowsum) {
  __shared__ float zsh[MCOLS];
  __shared__ float sgnsh[MCOLS];
  __shared__ float ranksh[MCOLS];
  __shared__ int   ladsh[MCOLS];
  __shared__ float red[256];

  const int i = blockIdx.x;
  const int t = threadIdx.x;
  const int labi = labels[i & 127];       // lab[i] = labels[i % 128]

  if (t < MCOLS) {
    const int col = t + (t >= i ? 1 : 0); // skip diagonal, masked_select order
    const int ld  = labi - labels[col & 127];
    zsh[t]   = z[i * NROWS + col];
    sgnsh[t] = (float)((ld > 0) - (ld < 0));
    ladsh[t] = (ld >= 0) ? ld : -ld;
  }
  __syncthreads();

  // ranks = argsort(argsort(lad)) with stable sort == stable counting-sort ranks
  if (t == 0) {
    int pre[16];
    int cnt[16];
    for (int v = 0; v < 16; ++v) { cnt[v] = 0; }
    for (int j = 0; j < MCOLS; ++j) cnt[ladsh[j]]++;
    int s = 0;
    for (int v = 0; v < 16; ++v) { pre[v] = s; s += cnt[v]; }
    for (int j = 0; j < MCOLS; ++j) { ranksh[j] = (float)pre[ladsh[j]]; pre[ladsh[j]]++; }
  }
  __syncthreads();

  float term = 0.0f;
  if (t < MCOLS) {
    const int   k    = t;
    const float zk   = zsh[k];
    const float rk   = ranksh[k];
    const int   ladk = ladsh[k];
    float possum = 0.0f;
    float negsum = 0.0f;
    for (int j = 0; j < MCOLS; ++j) {
      const float d = sgnsh[j] * (zsh[j] - zk);     // d[i,k,j]
      const bool pos = (ladsh[j] == ladk) && (j != k);
      if (pos) {
        const float ad = fabsf(d);
        const float w  = 2.0f / (1.0f + expf(-(ad - DELTA_F)));  // 2*sigmoid(ad-delta)
        possum = fmaf(ad, w, possum);
      } else {
        const float margin = (ranksh[j] - rk) * (1.0f / DELTA_F);
        const float w = 2.0f / (1.0f + expf(-(margin - d)));     // 2*sigmoid(margin-d)
        negsum = fmaf(expf(-d), w, negsum);
      }
    }
    term = -possum - logf(negsum);   // pos_logits - log(neg_logits)
  }
  red[t] = term;
  __syncthreads();
  for (int s = 128; s > 0; s >>= 1) {
    if (t < s) red[t] += red[t + s];
    __syncthreads();
  }
  if (t == 0) rowsum[i] = red[0];
}

// -------- kernel 4: final scalar --------
__global__ void __launch_bounds__(256) finalize_kernel(const float* __restrict__ rowsum,
                                                       float* __restrict__ out) {
  __shared__ float red[256];
  const int t = threadIdx.x;
  red[t] = rowsum[t];
  __syncthreads();
  for (int s = 128; s > 0; s >>= 1) {
    if (t < s) red[t] += red[t + s];
    __syncthreads();
  }
  if (t == 0) out[0] = -red[0] / (float)(NROWS * MCOLS);
}

extern "C" void kernel_launch(void* const* d_in, const int* in_sizes, int n_in,
                              void* d_out, int out_size, void* d_ws, size_t ws_size,
                              hipStream_t stream) {
  (void)in_sizes; (void)n_in; (void)out_size; (void)ws_size;
  const float* feats  = (const float*)d_in[0];   // [128,2,256] fp32
  const int*   labels = (const int*)d_in[1];     // [128,1] int32
  float* out = (float*)d_out;                    // scalar fp32

  float* z      = (float*)d_ws;                  // 256*256 floats (256 KB)
  float* norms  = z + NROWS * NROWS;             // 256 floats
  float* rowsum = norms + NROWS;                 // 256 floats

  norms_kernel  <<<1,     NROWS, 0, stream>>>(feats, norms);
  gram_z_kernel <<<256,   32,    0, stream>>>(feats, norms, z);
  row_loss_kernel<<<NROWS, 256,  0, stream>>>(z, labels, rowsum);
  finalize_kernel<<<1,    256,   0, stream>>>(rowsum, out);
}
